// Chatbot_84859963834395
// MI455X (gfx1250) — matrix operand order repaired and checked by reading, not compiled, not run on hardware
//
#include <hip/hip_runtime.h>
#include <hip/hip_bf16.h>
#include <stdint.h>

typedef unsigned short u16;
typedef __attribute__((ext_vector_type(16))) __bf16 bf16x16;
typedef __attribute__((ext_vector_type(8)))  float  f32x8;

#define B_  32
#define T_  512
#define V_  8000
#define E_  256
#define U_  512
#define G4_ 2048   // 4*U
#define D2_ 1024   // 2*U

union Frag { bf16x16 v; uint4 q[2]; };

__device__ __forceinline__ u16 f2bf(float f) {
  unsigned u = __float_as_uint(f);
  u += 0x7FFFu + ((u >> 16) & 1u);          // round-to-nearest-even
  return (u16)(u >> 16);
}
__device__ __forceinline__ float bf2f(u16 h) {
  return __uint_as_float(((unsigned)h) << 16);
}
__device__ __forceinline__ float sigmoidf(float x) {
  return 1.0f / (1.0f + __expf(-x));
}

// ---------------------------------------------------------------------------
// fp32 [K][N] row-major  ->  bf16 [N][K] (N-major, K contiguous)
// ---------------------------------------------------------------------------
__global__ void tconv_kernel(const float* __restrict__ in, u16* __restrict__ out,
                             int K, int N) {
  size_t i = (size_t)blockIdx.x * blockDim.x + threadIdx.x;
  size_t total = (size_t)K * N;
  if (i >= total) return;
  int k = (int)(i / N);
  int n = (int)(i % N);
  out[(size_t)n * K + k] = f2bf(in[i]);
}

// ---------------------------------------------------------------------------
// embedding gather: x_bf16[b,t,e] = bf16(emb[tokens[b,t], e])
// ---------------------------------------------------------------------------
__global__ void gather_kernel(const int* __restrict__ tokens,
                              const float* __restrict__ emb,
                              u16* __restrict__ x) {
  size_t i = (size_t)blockIdx.x * blockDim.x + threadIdx.x;
  size_t total = (size_t)B_ * T_ * E_;
  if (i >= total) return;
  int bt = (int)(i / E_);
  int e  = (int)(i % E_);
  int tok = tokens[bt];
  x[i] = f2bf(emb[(size_t)tok * E_ + e]);
}

// ---------------------------------------------------------------------------
// bf16 GEMM:  Out[M][N] = A[M][K] * Bt[N][K]^T + bias[N]
// Block = 256 thr = 8 waves, wave tile 16x64 (4 accumulators), block 32x256.
// Branch-free K-loop: ragged-N subtiles are CLAMPED to column N-16 (loads stay
// in-bounds, EXEC stays all-ones for WMMA); only the stores are guarded.
// ---------------------------------------------------------------------------
__global__ void gemm_bf16_kernel(const u16* __restrict__ A,
                                 const u16* __restrict__ Bt,
                                 const float* __restrict__ bias,
                                 void* __restrict__ Out,
                                 int M, int N, int K, int out_is_bf16) {
  const int lane = threadIdx.x & 31;
  const int wave = threadIdx.x >> 5;
  const int half = lane >> 4;
  const int l16  = lane & 15;
  const int mt   = wave & 1;
  const int nt   = wave >> 1;
  const int m0   = blockIdx.x * 32 + mt * 16;
  const int n0   = blockIdx.y * 256 + nt * 64;

  int nc[4];
#pragma unroll
  for (int s = 0; s < 4; ++s) {
    int n = n0 + s * 16;
    nc[s] = (n <= N - 16) ? n : (N - 16);   // clamp: always valid
  }

  const u16* ap = A + (size_t)(m0 + l16) * K + half * 16;
  const u16* bp0 = Bt + (size_t)(nc[0] + l16) * K + half * 16;
  const u16* bp1 = Bt + (size_t)(nc[1] + l16) * K + half * 16;
  const u16* bp2 = Bt + (size_t)(nc[2] + l16) * K + half * 16;
  const u16* bp3 = Bt + (size_t)(nc[3] + l16) * K + half * 16;

  f32x8 acc0 = {}, acc1 = {}, acc2 = {}, acc3 = {};
  for (int k0 = 0; k0 < K; k0 += 32) {
    Frag a;
    a.q[0] = *(const uint4*)(ap);
    a.q[1] = *(const uint4*)(ap + 8);
    __builtin_prefetch(bp0 + 32, 0, 3);
    Frag b;
    b.q[0] = *(const uint4*)(bp0); b.q[1] = *(const uint4*)(bp0 + 8);
    acc0 = __builtin_amdgcn_wmma_f32_16x16x32_bf16(false, a.v, false, b.v,
                                                   (short)0, acc0, false, false);
    b.q[0] = *(const uint4*)(bp1); b.q[1] = *(const uint4*)(bp1 + 8);
    acc1 = __builtin_amdgcn_wmma_f32_16x16x32_bf16(false, a.v, false, b.v,
                                                   (short)0, acc1, false, false);
    b.q[0] = *(const uint4*)(bp2); b.q[1] = *(const uint4*)(bp2 + 8);
    acc2 = __builtin_amdgcn_wmma_f32_16x16x32_bf16(false, a.v, false, b.v,
                                                   (short)0, acc2, false, false);
    b.q[0] = *(const uint4*)(bp3); b.q[1] = *(const uint4*)(bp3 + 8);
    acc3 = __builtin_amdgcn_wmma_f32_16x16x32_bf16(false, a.v, false, b.v,
                                                   (short)0, acc3, false, false);
    ap += 32; bp0 += 32; bp1 += 32; bp2 += 32; bp3 += 32;
  }

  f32x8 accs[4] = { acc0, acc1, acc2, acc3 };
#pragma unroll
  for (int s = 0; s < 4; ++s) {
    if (n0 + s * 16 >= N) continue;         // store guard only (after all WMMA)
    int n = n0 + s * 16 + l16;
    float bv = bias ? bias[n] : 0.0f;
#pragma unroll
    for (int v = 0; v < 8; ++v) {
      int m = m0 + v + 8 * half;
      float val = accs[s][v] + bv;
      if (out_is_bf16) ((u16*)Out)[(size_t)m * N + n] = f2bf(val);
      else             ((float*)Out)[(size_t)m * N + n] = val;
    }
  }
}

// ---------------------------------------------------------------------------
// One fused LSTM time step, both directions. 16 blocks x 256 threads.
//   block: dir = bid>>3, ublock = bid&7 (64 u-columns)
//   wave:  mt = batch-tile (0..1), ng = 16-col group (0..3)
//   wave computes the 4 gate tiles (i,f,g,o) of the SAME 16 columns -> gate
//   nonlinearity entirely in registers, no LDS, no barrier, no divergence.
// h-state double-buffered on step parity (cross-block RAW on full-K reads).
// ---------------------------------------------------------------------------
__global__ void lstm_step_kernel(const int* __restrict__ tokens,
                                 const u16* __restrict__ rtf,   // [2048][512] bf16
                                 const u16* __restrict__ rtb,
                                 const u16* __restrict__ xwf,   // [B*T][2048] bf16
                                 const u16* __restrict__ xwb,
                                 u16* __restrict__ hstate,      // [2][2][B][U] bf16
                                 float* __restrict__ cstate,    // [2][B][U]
                                 u16* __restrict__ hout,        // [B*T][2U] bf16
                                 int step) {
  const int dir = blockIdx.x >> 3;
  const int ub  = blockIdx.x & 7;
  const int t   = dir ? (T_ - 1 - step) : step;
  const int lane = threadIdx.x & 31;
  const int wave = threadIdx.x >> 5;
  const int half = lane >> 4;
  const int l16  = lane & 15;
  const int mt   = wave & 1;
  const int ng   = wave >> 1;
  const int u0   = ub * 64 + ng * 16;

  const u16* rt = dir ? rtb : rtf;
  const u16* xw = dir ? xwb : xwf;
  const int rb = step & 1;
  const u16* hsR = hstate + (size_t)dir * 2 * B_ * U_ + (size_t)rb * B_ * U_;
  u16*       hsW = hstate + (size_t)dir * 2 * B_ * U_ + (size_t)(rb ^ 1) * B_ * U_;
  float*     cs  = cstate + (size_t)dir * B_ * U_;

  const u16* ap  = hsR + (size_t)(mt * 16 + l16) * U_ + half * 16;
  const u16* bpi = rt + (size_t)(0 * U_ + u0 + l16) * U_ + half * 16;
  const u16* bpf = rt + (size_t)(1 * U_ + u0 + l16) * U_ + half * 16;
  const u16* bpg = rt + (size_t)(2 * U_ + u0 + l16) * U_ + half * 16;
  const u16* bpo = rt + (size_t)(3 * U_ + u0 + l16) * U_ + half * 16;

  f32x8 ai = {}, af = {}, ag = {}, ao = {};
  for (int k0 = 0; k0 < U_; k0 += 32) {
    Frag a;
    a.q[0] = *(const uint4*)(ap);
    a.q[1] = *(const uint4*)(ap + 8);
    Frag b;
    b.q[0] = *(const uint4*)(bpi); b.q[1] = *(const uint4*)(bpi + 8);
    ai = __builtin_amdgcn_wmma_f32_16x16x32_bf16(false, a.v, false, b.v,
                                                 (short)0, ai, false, false);
    b.q[0] = *(const uint4*)(bpf); b.q[1] = *(const uint4*)(bpf + 8);
    af = __builtin_amdgcn_wmma_f32_16x16x32_bf16(false, a.v, false, b.v,
                                                 (short)0, af, false, false);
    b.q[0] = *(const uint4*)(bpg); b.q[1] = *(const uint4*)(bpg + 8);
    ag = __builtin_amdgcn_wmma_f32_16x16x32_bf16(false, a.v, false, b.v,
                                                 (short)0, ag, false, false);
    b.q[0] = *(const uint4*)(bpo); b.q[1] = *(const uint4*)(bpo + 8);
    ao = __builtin_amdgcn_wmma_f32_16x16x32_bf16(false, a.v, false, b.v,
                                                 (short)0, ao, false, false);
    ap += 32; bpi += 32; bpf += 32; bpg += 32; bpo += 32;
  }

  const int u = u0 + l16;
#pragma unroll
  for (int v = 0; v < 8; ++v) {
    const int b = mt * 16 + v + 8 * half;           // batch row (C layout)
    const size_t row = (size_t)b * T_ + t;
    const float zi = ai[v] + bf2f(xw[row * G4_ + 0 * U_ + u]);
    const float zf = af[v] + bf2f(xw[row * G4_ + 1 * U_ + u]);
    const float zg = ag[v] + bf2f(xw[row * G4_ + 2 * U_ + u]);
    const float zo = ao[v] + bf2f(xw[row * G4_ + 3 * U_ + u]);
    const float c_old = cs[(size_t)b * U_ + u];
    const float h_old = bf2f(hsR[(size_t)b * U_ + u]);
    float c_new = sigmoidf(zf) * c_old + sigmoidf(zi) * tanhf(zg);
    float h_new = sigmoidf(zo) * tanhf(c_new);
    if (tokens[row] == 0) { c_new = c_old; h_new = h_old; }   // Keras mask carry
    cs[(size_t)b * U_ + u]  = c_new;
    hsW[(size_t)b * U_ + u] = f2bf(h_new);
    hout[row * D2_ + dir * U_ + u] = f2bf(h_new);
  }
}

// ---------------------------------------------------------------------------
extern "C" void kernel_launch(void* const* d_in, const int* in_sizes, int n_in,
                              void* d_out, int out_size, void* d_ws, size_t ws_size,
                              hipStream_t stream) {
  const int*   tokens = (const int*)  d_in[0];
  const float* emb    = (const float*)d_in[1];
  const float* k1f = (const float*)d_in[2],  *r1f = (const float*)d_in[3],  *b1f = (const float*)d_in[4];
  const float* k1b = (const float*)d_in[5],  *r1b = (const float*)d_in[6],  *b1b = (const float*)d_in[7];
  const float* k2f = (const float*)d_in[8],  *r2f = (const float*)d_in[9],  *b2f = (const float*)d_in[10];
  const float* k2b = (const float*)d_in[11], *r2b = (const float*)d_in[12], *b2b = (const float*)d_in[13];
  const float* wd  = (const float*)d_in[14], *bd  = (const float*)d_in[15];
  float* out = (float*)d_out;

  // ---- workspace carving (256B aligned regions) ----
  char* p = (char*)d_ws;
  auto a_u16 = [&](size_t n) -> u16*   { u16*   r = (u16*)p;   p += ((n * 2 + 255) / 256) * 256; return r; };
  auto a_f32 = [&](size_t n) -> float* { float* r = (float*)p; p += ((n * 4 + 255) / 256) * 256; return r; };

  u16* k1f_t = a_u16((size_t)G4_ * E_);    u16* r1f_t = a_u16((size_t)G4_ * U_);
  u16* k1b_t = a_u16((size_t)G4_ * E_);    u16* r1b_t = a_u16((size_t)G4_ * U_);
  u16* k2f_t = a_u16((size_t)G4_ * D2_);   u16* r2f_t = a_u16((size_t)G4_ * U_);
  u16* k2b_t = a_u16((size_t)G4_ * D2_);   u16* r2b_t = a_u16((size_t)G4_ * U_);
  u16* wd_t  = a_u16((size_t)V_ * D2_);
  u16* xbf   = a_u16((size_t)B_ * T_ * E_);
  u16* h1bf  = a_u16((size_t)B_ * T_ * D2_);
  u16* h2bf  = a_u16((size_t)B_ * T_ * D2_);
  u16* xwf   = a_u16((size_t)B_ * T_ * G4_);
  u16* xwb   = a_u16((size_t)B_ * T_ * G4_);
  u16*   hstate = a_u16((size_t)2 * 2 * B_ * U_);
  float* cstate = a_f32((size_t)2 * B_ * U_);

  auto tconv = [&](const float* in, u16* o, int K, int N) {
    size_t tot = (size_t)K * N;
    tconv_kernel<<<(unsigned)((tot + 255) / 256), 256, 0, stream>>>(in, o, K, N);
  };
  auto gemm = [&](const u16* A, const u16* Bt, const float* bias, void* O,
                  int M, int N, int K, int obf) {
    dim3 grid(M / 32, (N + 255) / 256);
    gemm_bf16_kernel<<<grid, 256, 0, stream>>>(A, Bt, bias, O, M, N, K, obf);
  };

  // 1. weight convert + transpose (fp32 [K][N] -> bf16 [N][K])
  tconv(k1f, k1f_t, E_,  G4_);  tconv(r1f, r1f_t, U_, G4_);
  tconv(k1b, k1b_t, E_,  G4_);  tconv(r1b, r1b_t, U_, G4_);
  tconv(k2f, k2f_t, D2_, G4_);  tconv(r2f, r2f_t, U_, G4_);
  tconv(k2b, k2b_t, D2_, G4_);  tconv(r2b, r2b_t, U_, G4_);
  tconv(wd,  wd_t,  D2_, V_);

  // 2. embedding gather -> bf16 activations
  {
    size_t tot = (size_t)B_ * T_ * E_;
    gather_kernel<<<(unsigned)((tot + 255) / 256), 256, 0, stream>>>(tokens, emb, xbf);
  }

  const int M = B_ * T_;

  // 3. layer-1 input projections (+bias), bf16 out
  gemm(xbf, k1f_t, b1f, xwf, M, G4_, E_, 1);
  gemm(xbf, k1b_t, b1b, xwb, M, G4_, E_, 1);

  // 4. layer-1 bidirectional scan
  hipMemsetAsync(hstate, 0, (size_t)2 * 2 * B_ * U_ * sizeof(u16), stream);
  hipMemsetAsync(cstate, 0, (size_t)2 * B_ * U_ * sizeof(float), stream);
  for (int s = 0; s < T_; ++s)
    lstm_step_kernel<<<16, 256, 0, stream>>>(tokens, r1f_t, r1b_t, xwf, xwb,
                                             hstate, cstate, h1bf, s);

  // 5. layer-2 input projections (reuse xW buffers)
  gemm(h1bf, k2f_t, b2f, xwf, M, G4_, D2_, 1);
  gemm(h1bf, k2b_t, b2b, xwb, M, G4_, D2_, 1);

  // 6. layer-2 bidirectional scan
  hipMemsetAsync(hstate, 0, (size_t)2 * 2 * B_ * U_ * sizeof(u16), stream);
  hipMemsetAsync(cstate, 0, (size_t)2 * B_ * U_ * sizeof(float), stream);
  for (int s = 0; s < T_; ++s)
    lstm_step_kernel<<<16, 256, 0, stream>>>(tokens, r2f_t, r2b_t, xwf, xwb,
                                             hstate, cstate, h2bf, s);

  // 7. output dense: logits = h2 * wd^T + bd   (fp32 out, N=8000 guarded)
  gemm(h2bf, wd_t, bd, out, M, V_, D2_, 0);
}
